// SemanticAlignmentModule_78030965833885
// MI455X (gfx1250) — compile-verified, matrix operand order checked by reading, not run on hardware
//
#include <hip/hip_runtime.h>
#include <hip/hip_bf16.h>
#include <math.h>

typedef __attribute__((ext_vector_type(16))) _Float16 v16h;
typedef __attribute__((ext_vector_type(8)))  _Float16 v8h;
typedef __attribute__((ext_vector_type(8)))  float    v8f;

#define ROWS   32768   // B*H*W = 8*64*64
#define CDIM   512
#define TROWS  616     // B*T = 8*77
#define TLEN   77
#define FDIM   2048
#define NHEAD  8
#define DH     64

// ---------------------------------------------------------------------------
// helpers
// ---------------------------------------------------------------------------
__device__ __forceinline__ v8f zero8() {
  v8f z = {0.f,0.f,0.f,0.f,0.f,0.f,0.f,0.f};
  return z;
}

__device__ __forceinline__ v16h make_frag(v8h lo, v8h hi) {
  v16h o;
#pragma unroll
  for (int i = 0; i < 8; ++i) { o[i] = lo[i]; o[i + 8] = hi[i]; }
  return o;
}

// A fragment: 16x32 f16 tile at (row0, k0) of row-major [M x ld], rows clamped.
// ISA layout: lanes 0-15 -> M=lane, K = k0+0..7 / k0+16..23
//             lanes 16-31 -> M=lane-16, K = k0+8..15 / k0+24..31
__device__ __forceinline__ v16h frag_load_a(const _Float16* base, int row0,
                                            int ld, int k0, int M) {
  int lane = threadIdx.x & 31;
  int r = row0 + (lane & 15);
  r = (r < M) ? r : (M - 1);
  int kk = k0 + ((lane >> 4) << 3);
  const _Float16* p = base + (size_t)r * ld + kk;
  v8h lo = *(const v8h*)(p);
  v8h hi = *(const v8h*)(p + 16);
  return make_frag(lo, hi);
}

// B fragment from a transposed weight WT [N x K] row-major: column n0 of the
// logical K x N matrix == contiguous row of WT. Symmetric lane striping.
__device__ __forceinline__ v16h frag_load_b(const _Float16* wt, int n0,
                                            int ld, int k0) {
  int lane = threadIdx.x & 31;
  int n = n0 + (lane & 15);
  int kk = k0 + ((lane >> 4) << 3);
  const _Float16* p = wt + (size_t)n * ld + kk;
  v8h lo = *(const v8h*)(p);
  v8h hi = *(const v8h*)(p + 16);
  return make_frag(lo, hi);
}

// A/B fragment from an LDS tile of leading dimension ld (halves)
__device__ __forceinline__ v16h frag_load_lds(const _Float16* buf, int ld,
                                              int k0) {
  int lane = threadIdx.x & 31;
  int r = lane & 15;
  int kk = k0 + ((lane >> 4) << 3);
  const _Float16* p = buf + r * ld + kk;
  v8h lo = *(const v8h*)(p);
  v8h hi = *(const v8h*)(p + 16);
  return make_frag(lo, hi);
}

// generic(shared) pointer -> 32-bit LDS byte offset (flat aperture rule:
// offset is the low 32 bits of the address, ISA 10.2)
__device__ __forceinline__ unsigned lds_off(const void* p) {
  return (unsigned)(uintptr_t)p;
}

// async copy of 16 bytes global -> LDS, tracked with ASYNCcnt (no VGPRs)
__device__ __forceinline__ void async_copy_b128(unsigned ldsaddr,
                                                const void* gaddr) {
  asm volatile("global_load_async_to_lds_b128 %0, %1, off"
               :
               : "v"(ldsaddr), "v"((unsigned long long)(uintptr_t)gaddr)
               : "memory");
}

__device__ __forceinline__ void wait_asynccnt0() {
  asm volatile("s_wait_asynccnt 0" ::: "memory");
}

__device__ __forceinline__ float block_reduce_sum_256(float v) {
  __shared__ float sbuf[256];
  int t = threadIdx.x;
  sbuf[t] = v;
  __syncthreads();
#pragma unroll
  for (int s = 128; s > 0; s >>= 1) {
    if (t < s) sbuf[t] += sbuf[t + s];
    __syncthreads();
  }
  float r = sbuf[0];
  __syncthreads();
  return r;
}

// ---------------------------------------------------------------------------
// prep kernels
// ---------------------------------------------------------------------------
__global__ void transpose_f32_to_f16(const float* __restrict__ src,
                                     _Float16* __restrict__ dst,
                                     int K, int N) {
  int idx = blockIdx.x * blockDim.x + threadIdx.x;
  if (idx >= K * N) return;
  int k = idx / N, n = idx - k * N;
  dst[(size_t)n * K + k] = (_Float16)src[idx];
}

__global__ void convert_f32_to_f16(const float* __restrict__ src,
                                   _Float16* __restrict__ dst, int n) {
  int idx = blockIdx.x * blockDim.x + threadIdx.x;
  if (idx < n) dst[idx] = (_Float16)src[idx];
}

// LayerNorm over last dim (512), one block of 256 per row -> f32 + f16 copies
__global__ void ln_rows(const float* __restrict__ in,
                        const float* __restrict__ w,
                        const float* __restrict__ b,
                        float* __restrict__ of32,
                        _Float16* __restrict__ of16) {
  int row = blockIdx.x, t = threadIdx.x;
  const float* p = in + (size_t)row * CDIM;
  float a0 = p[t], a1 = p[t + 256];
  float mean = block_reduce_sum_256(a0 + a1) * (1.f / CDIM);
  float d0 = a0 - mean, d1 = a1 - mean;
  float var = block_reduce_sum_256(d0 * d0 + d1 * d1) * (1.f / CDIM);
  float inv = rsqrtf(var + 1e-5f);
  float y0 = d0 * inv * w[t] + b[t];
  float y1 = d1 * inv * w[t + 256] + b[t + 256];
  of32[(size_t)row * CDIM + t] = y0;
  of32[(size_t)row * CDIM + t + 256] = y1;
  of16[(size_t)row * CDIM + t] = (_Float16)y0;
  of16[(size_t)row * CDIM + t + 256] = (_Float16)y1;
}

// ---------------------------------------------------------------------------
// generic WMMA GEMM: C[MxN] = A[MxK](f16,rowmaj) * W[KxN] (given as WT[NxK]
// f16 rowmaj) + bias, writing optional f32 and/or f16 outputs.
// Block = 256 threads = 8 waves arranged 2(M) x 4(N); wave tile 32x32.
// ---------------------------------------------------------------------------
__global__ void gemm_wmma(const _Float16* __restrict__ A,
                          const _Float16* __restrict__ BT,
                          const float* __restrict__ bias,
                          float* __restrict__ Cf32,
                          _Float16* __restrict__ Cf16,
                          int M, int N, int K) {
  int lane = threadIdx.x & 31;
  int wave = threadIdx.x >> 5;
  int wm = wave >> 2, wn = wave & 3;
  int rowBase = blockIdx.y * 64 + wm * 32;
  int colBase = blockIdx.x * 128 + wn * 32;

  v8f acc00 = zero8(), acc01 = zero8(), acc10 = zero8(), acc11 = zero8();

  for (int k0 = 0; k0 < K; k0 += 32) {
    // hint the streaming A operand into cache ahead of time
    if (k0 + 64 < K) {
      int rp = rowBase + (lane & 15);
      rp = (rp < M) ? rp : (M - 1);
      __builtin_prefetch(A + (size_t)rp * K + k0 + 64, 0, 1);
    }
    v16h a0 = frag_load_a(A, rowBase, K, k0, M);
    v16h a1 = frag_load_a(A, rowBase + 16, K, k0, M);
    v16h b0 = frag_load_b(BT, colBase, K, k0);
    v16h b1 = frag_load_b(BT, colBase + 16, K, k0);
    acc00 = __builtin_amdgcn_wmma_f32_16x16x32_f16(false, a0, false, b0,
                                                   (short)0, acc00, false, false);
    acc01 = __builtin_amdgcn_wmma_f32_16x16x32_f16(false, a0, false, b1,
                                                   (short)0, acc01, false, false);
    acc10 = __builtin_amdgcn_wmma_f32_16x16x32_f16(false, a1, false, b0,
                                                   (short)0, acc10, false, false);
    acc11 = __builtin_amdgcn_wmma_f32_16x16x32_f16(false, a1, false, b1,
                                                   (short)0, acc11, false, false);
  }

  // C/D layout: VGPR j -> (M = j + (lane>=16 ? 8 : 0), N = lane&15)
  int nloc = lane & 15;
  int msub = (lane >> 4) * 8;
#pragma unroll
  for (int i = 0; i < 2; ++i) {
#pragma unroll
    for (int j = 0; j < 2; ++j) {
      v8f acc = (i == 0) ? ((j == 0) ? acc00 : acc01)
                         : ((j == 0) ? acc10 : acc11);
      int col = colBase + j * 16 + nloc;
      float bval = bias ? bias[col] : 0.f;
#pragma unroll
      for (int r = 0; r < 8; ++r) {
        int row = rowBase + i * 16 + msub + r;
        if (row < M) {
          float val = acc[r] + bval;
          if (Cf32) Cf32[(size_t)row * N + col] = val;
          if (Cf16) Cf16[(size_t)row * N + col] = (_Float16)val;
        }
      }
    }
  }
}

// ---------------------------------------------------------------------------
// small per-row kernels
// ---------------------------------------------------------------------------
__global__ void pad_mask_kernel(const float* __restrict__ text,
                                float* __restrict__ mask) {
  int row = blockIdx.x, t = threadIdx.x;
  const float* p = text + (size_t)row * CDIM;
  float s = fabsf(p[t]) + fabsf(p[t + 256]);
  float tot = block_reduce_sum_256(s);
  if (t == 0) mask[row] = (tot <= 1e-6f) ? 1.f : 0.f;
}

__global__ void l2norm_rows_f16(_Float16* __restrict__ x) {
  int row = blockIdx.x, t = threadIdx.x;
  _Float16* p = x + (size_t)row * CDIM;
  float a0 = (float)p[t], a1 = (float)p[t + 256];
  float ss = block_reduce_sum_256(a0 * a0 + a1 * a1);
  float inv = 1.f / fmaxf(sqrtf(ss), 1e-6f);
  p[t] = (_Float16)(a0 * inv);
  p[t + 256] = (_Float16)(a1 * inv);
}

// ---------------------------------------------------------------------------
// attention with top-5 masking: one thread per (b, n, head). T=77 keys.
// streaming top-5 threshold then one-pass softmax * V accumulation.
// ---------------------------------------------------------------------------
__global__ void attention_topk(const _Float16* __restrict__ q,
                               const _Float16* __restrict__ k,
                               const _Float16* __restrict__ v,
                               const float* __restrict__ mask,
                               const float* __restrict__ logit_scale_p,
                               _Float16* __restrict__ alignedOut) {
  int idx = blockIdx.x * blockDim.x + threadIdx.x;
  int h = idx & (NHEAD - 1);
  int bn = idx >> 3;          // 0..32767
  int b = bn >> 12;           // /4096
  float ls = logit_scale_p[0];
  ls = fminf(fmaxf(ls, -2.f), 2.f);
  float scale = expf(ls) * 0.125f;   // 1/sqrt(64)

  const _Float16* qp = q + (size_t)bn * CDIM + h * DH;
  float qr[DH];
#pragma unroll
  for (int i = 0; i < DH; ++i) qr[i] = (float)qp[i];

  const _Float16* kbase = k + ((size_t)b * TLEN) * CDIM + h * DH;
  const _Float16* vbase = v + ((size_t)b * TLEN) * CDIM + h * DH;
  const float* mrow = mask + b * TLEN;

  // pass 1: streaming top-5 (t0 >= t1 >= ... >= t4)
  float t0 = -1e30f, t1 = -1e30f, t2 = -1e30f, t3 = -1e30f, t4 = -1e30f;
  for (int t = 0; t < TLEN; ++t) {
    if (mrow[t] > 0.5f) continue;
    const _Float16* kp = kbase + (size_t)t * CDIM;
    float s = 0.f;
    for (int i = 0; i < DH; ++i) s += qr[i] * (float)kp[i];
    s *= scale;
    if (s > t4) t4 = s;
    if (t4 > t3) { float x = t3; t3 = t4; t4 = x; }
    if (t3 > t2) { float x = t2; t2 = t3; t3 = x; }
    if (t2 > t1) { float x = t1; t1 = t2; t2 = x; }
    if (t1 > t0) { float x = t0; t0 = t1; t1 = x; }
  }
  float kth = t4, mx = t0;

  // pass 2: softmax over kept entries, weighted V accumulation
  float acc[DH];
#pragma unroll
  for (int i = 0; i < DH; ++i) acc[i] = 0.f;
  float denom = 0.f;
  for (int t = 0; t < TLEN; ++t) {
    if (mrow[t] > 0.5f) continue;
    const _Float16* kp = kbase + (size_t)t * CDIM;
    float s = 0.f;
    for (int i = 0; i < DH; ++i) s += qr[i] * (float)kp[i];
    s *= scale;
    if (s < kth) continue;
    float w = expf(s - mx);
    denom += w;
    const _Float16* vp = vbase + (size_t)t * CDIM;
    for (int i = 0; i < DH; ++i) acc[i] += w * (float)vp[i];
  }
  float inv = 1.f / denom;
  _Float16* op = alignedOut + (size_t)bn * CDIM + h * DH;
#pragma unroll
  for (int i = 0; i < DH; ++i) op[i] = (_Float16)(acc[i] * inv);
}

// ---------------------------------------------------------------------------
// gate + residual + LayerNorm2: one block per row
// ---------------------------------------------------------------------------
__global__ void residual_gate_ln2(const float* __restrict__ x,
                                  const float* __restrict__ alignedO,
                                  const float* __restrict__ gate_w,
                                  const float* __restrict__ gate_b,
                                  const float* __restrict__ alpha_p,
                                  const float* __restrict__ w,
                                  const float* __restrict__ b,
                                  float* __restrict__ yf32,
                                  _Float16* __restrict__ yf16) {
  int row = blockIdx.x, t = threadIdx.x;
  const float* xp = x + (size_t)row * CDIM;
  const float* ap = alignedO + (size_t)row * CDIM;
  float x0 = xp[t], x1 = xp[t + 256];
  float a0 = ap[t], a1 = ap[t + 256];
  float gdot = block_reduce_sum_256(x0 * gate_w[t] + x1 * gate_w[t + 256]);
  float gate = 1.f / (1.f + expf(-(gdot + gate_b[0])));
  float am = alpha_p[0] * gate;
  float y0 = x0 + am * a0;
  float y1 = x1 + am * a1;
  float mean = block_reduce_sum_256(y0 + y1) * (1.f / CDIM);
  float d0 = y0 - mean, d1 = y1 - mean;
  float var = block_reduce_sum_256(d0 * d0 + d1 * d1) * (1.f / CDIM);
  float inv = rsqrtf(var + 1e-5f);
  float o0 = d0 * inv * w[t] + b[t];
  float o1 = d1 * inv * w[t + 256] + b[t + 256];
  yf32[(size_t)row * CDIM + t] = o0;
  yf32[(size_t)row * CDIM + t + 256] = o1;
  yf16[(size_t)row * CDIM + t] = (_Float16)o0;
  yf16[(size_t)row * CDIM + t + 256] = (_Float16)o1;
}

// ---------------------------------------------------------------------------
// fused FFN: per 16-row block.
// stage 0: async-copy the whole 16x512 y tile (16KB) into LDS once
//          (GLOBAL_LOAD_ASYNC_TO_LDS_B128, ASYNCcnt) -- removes the 8-wave x
//          16-chunk redundant global re-reads of y.
// then 16 chunks of 128 hidden columns:
//   phase1: each of 8 waves computes one 16x16 h tile (WMMA over K=512 from
//           LDS), applies exact GELU, stages f16 into LDS.
//   phase2: all waves consume LDS h as A-fragments against W2T, accumulating
//           a 16x512 output entirely in registers. h never touches HBM.
// ---------------------------------------------------------------------------
__global__ void ffn_fused(const _Float16* __restrict__ yf16,
                          const float* __restrict__ yf32,
                          const _Float16* __restrict__ W1T,
                          const float* __restrict__ b1,
                          const _Float16* __restrict__ W2T,
                          const float* __restrict__ b2,
                          float* __restrict__ out, int M) {
  __shared__ __align__(16) _Float16 ybuf[16 * 512];   // 16 KB input tile
  __shared__ __align__(16) _Float16 hbuf[16 * 128];   // 4 KB hidden chunk
  int lane = threadIdx.x & 31;
  int wave = threadIdx.x >> 5;
  int rowBase = blockIdx.x * 16;

  // ---- async stage of the y tile: 1024 x 16B chunks, 4 per thread ----
  {
    int tid = threadIdx.x;
#pragma unroll
    for (int i = 0; i < 4; ++i) {
      int c = tid + i * 256;          // 0..1023
      int row = c >> 6;               // 64 x 16B chunks per 1KB row
      int seg = c & 63;
      const _Float16* g = yf16 + (size_t)(rowBase + row) * CDIM + seg * 8;
      async_copy_b128(lds_off(&ybuf[row * CDIM + seg * 8]), g);
    }
    wait_asynccnt0();
    __syncthreads();
  }

  v8f oacc[4];
#pragma unroll
  for (int j = 0; j < 4; ++j) oacc[j] = zero8();

  int nloc = lane & 15;
  int msub = (lane >> 4) * 8;

  for (int chunk = 0; chunk < 16; ++chunk) {
    int c0 = chunk * 128;
    __syncthreads();   // protect hbuf from previous chunk's phase 2 readers

    // phase 1: h tile for hidden columns [c0 + wave*16, +16)
    int n1 = c0 + wave * 16;
    v8f hacc = zero8();
    for (int k0 = 0; k0 < CDIM; k0 += 32) {
      v16h a = frag_load_lds(ybuf, CDIM, k0);
      v16h b = frag_load_b(W1T, n1, CDIM, k0);
      hacc = __builtin_amdgcn_wmma_f32_16x16x32_f16(false, a, false, b,
                                                    (short)0, hacc, false, false);
    }
#pragma unroll
    for (int r = 0; r < 8; ++r) {
      float hv = hacc[r] + b1[n1 + nloc];
      hv = 0.5f * hv * (1.f + erff(hv * 0.70710678f));   // exact GELU
      hbuf[(msub + r) * 128 + wave * 16 + nloc] = (_Float16)hv;
    }
    __syncthreads();

    // phase 2: oacc(16 x 64 per wave) += h(16 x 128) * W2[c0:c0+128, :]
    for (int k2 = 0; k2 < 128; k2 += 32) {
      v16h a = frag_load_lds(hbuf, 128, k2);
#pragma unroll
      for (int j = 0; j < 4; ++j) {
        int n2 = wave * 64 + j * 16;
        v16h b = frag_load_b(W2T, n2, FDIM, c0 + k2);
        oacc[j] = __builtin_amdgcn_wmma_f32_16x16x32_f16(false, a, false, b,
                                                         (short)0, oacc[j],
                                                         false, false);
      }
    }
  }

  // epilogue: residual + bias
#pragma unroll
  for (int j = 0; j < 4; ++j) {
    int col = wave * 64 + j * 16 + nloc;
    float bv = b2[col];
#pragma unroll
    for (int r = 0; r < 8; ++r) {
      int row = rowBase + msub + r;
      if (row < M)
        out[(size_t)row * CDIM + col] =
            yf32[(size_t)row * CDIM + col] + oacc[j][r] + bv;
    }
  }
}

// ---------------------------------------------------------------------------
// launch
// ---------------------------------------------------------------------------
extern "C" void kernel_launch(void* const* d_in, const int* in_sizes, int n_in,
                              void* d_out, int out_size, void* d_ws, size_t ws_size,
                              hipStream_t stream) {
  const float* visual = (const float*)d_in[0];
  const float* text   = (const float*)d_in[1];
  const float* n1w    = (const float*)d_in[2];
  const float* n1b    = (const float*)d_in[3];
  const float* Wq     = (const float*)d_in[4];
  const float* bq     = (const float*)d_in[5];
  const float* Wk     = (const float*)d_in[6];
  const float* bk     = (const float*)d_in[7];
  const float* Wv     = (const float*)d_in[8];
  const float* bv     = (const float*)d_in[9];
  const float* Wo     = (const float*)d_in[10];
  const float* bo     = (const float*)d_in[11];
  const float* gate_w = (const float*)d_in[12];
  const float* gate_b = (const float*)d_in[13];
  const float* n2w    = (const float*)d_in[14];
  const float* n2b    = (const float*)d_in[15];
  const float* W1     = (const float*)d_in[16];
  const float* b1     = (const float*)d_in[17];
  const float* W2     = (const float*)d_in[18];
  const float* b2     = (const float*)d_in[19];
  const float* logit_scale = (const float*)d_in[20];
  const float* alpha       = (const float*)d_in[21];
  float* out = (float*)d_out;

  // workspace carve-out (256B aligned). aliasing:
  //   xf16 buffer is reused for aligned(f16); qf16 buffer is reused for y(f16)
  char* base = (char*)d_ws;
  size_t off = 0;
  auto carve = [&](size_t bytes) -> char* {
    char* p = base + off;
    off = (off + bytes + 255) & ~(size_t)255;
    return p;
  };
  float*    xf32     = (float*)   carve((size_t)ROWS * CDIM * 4);
  _Float16* xf16     = (_Float16*)carve((size_t)ROWS * CDIM * 2);  // -> aligned f16
  _Float16* qf16     = (_Float16*)carve((size_t)ROWS * CDIM * 2);  // -> y f16
  float*    alignedO = (float*)   carve((size_t)ROWS * CDIM * 4);
  float*    yf32     = (float*)   carve((size_t)ROWS * CDIM * 4);
  _Float16* textf16  = (_Float16*)carve((size_t)TROWS * CDIM * 2);
  _Float16* kf16     = (_Float16*)carve((size_t)TROWS * CDIM * 2);
  _Float16* vf16     = (_Float16*)carve((size_t)TROWS * CDIM * 2);
  float*    maskf    = (float*)   carve((size_t)TROWS * 4);
  _Float16* WqT      = (_Float16*)carve((size_t)CDIM * CDIM * 2);
  _Float16* WkT      = (_Float16*)carve((size_t)CDIM * CDIM * 2);
  _Float16* WvT      = (_Float16*)carve((size_t)CDIM * CDIM * 2);
  _Float16* WoT      = (_Float16*)carve((size_t)CDIM * CDIM * 2);
  _Float16* W1T      = (_Float16*)carve((size_t)CDIM * FDIM * 2);
  _Float16* W2T      = (_Float16*)carve((size_t)FDIM * CDIM * 2);
  (void)ws_size; (void)in_sizes; (void)n_in; (void)out_size;

  // 1) weight prep (f32 -> transposed f16), text -> f16
  {
    int n = CDIM * CDIM, g = (n + 255) / 256;
    transpose_f32_to_f16<<<g, 256, 0, stream>>>(Wq, WqT, CDIM, CDIM);
    transpose_f32_to_f16<<<g, 256, 0, stream>>>(Wk, WkT, CDIM, CDIM);
    transpose_f32_to_f16<<<g, 256, 0, stream>>>(Wv, WvT, CDIM, CDIM);
    transpose_f32_to_f16<<<g, 256, 0, stream>>>(Wo, WoT, CDIM, CDIM);
    int n1e = CDIM * FDIM, g1 = (n1e + 255) / 256;
    transpose_f32_to_f16<<<g1, 256, 0, stream>>>(W1, W1T, CDIM, FDIM);
    transpose_f32_to_f16<<<g1, 256, 0, stream>>>(W2, W2T, FDIM, CDIM);
    int nt = TROWS * CDIM;
    convert_f32_to_f16<<<(nt + 255) / 256, 256, 0, stream>>>(text, textf16, nt);
  }

  // 2) LayerNorm1
  ln_rows<<<ROWS, 256, 0, stream>>>(visual, n1w, n1b, xf32, xf16);

  // 3) projections (WMMA)
  gemm_wmma<<<dim3(CDIM / 128, ROWS / 64), 256, 0, stream>>>(
      xf16, WqT, bq, nullptr, qf16, ROWS, CDIM, CDIM);
  gemm_wmma<<<dim3(CDIM / 128, (TROWS + 63) / 64), 256, 0, stream>>>(
      textf16, WkT, bk, nullptr, kf16, TROWS, CDIM, CDIM);
  gemm_wmma<<<dim3(CDIM / 128, (TROWS + 63) / 64), 256, 0, stream>>>(
      textf16, WvT, bv, nullptr, vf16, TROWS, CDIM, CDIM);

  // 4) pad mask + L2 norms (in place on f16)
  pad_mask_kernel<<<TROWS, 256, 0, stream>>>(text, maskf);
  l2norm_rows_f16<<<ROWS, 256, 0, stream>>>(qf16);
  l2norm_rows_f16<<<TROWS, 256, 0, stream>>>(kf16);

  // 5) top-5 cross attention -> aligned f16 (reuses xf16 buffer)
  attention_topk<<<(ROWS * NHEAD) / 256, 256, 0, stream>>>(
      qf16, kf16, vf16, maskf, logit_scale, xf16);

  // 6) output projection (WMMA)
  gemm_wmma<<<dim3(CDIM / 128, ROWS / 64), 256, 0, stream>>>(
      xf16, WoT, bo, alignedO, nullptr, ROWS, CDIM, CDIM);

  // 7) gate + residual + LayerNorm2 (y f16 reuses qf16 buffer)
  residual_gate_ln2<<<ROWS, 256, 0, stream>>>(
      xf32, alignedO, gate_w, gate_b, alpha, n2w, n2b, yf32, qf16);

  // 8) fused FFN + final residual -> d_out
  ffn_fused<<<ROWS / 16, 256, 0, stream>>>(
      qf16, yf32, W1T, b1, W2T, b2, out, ROWS);
}
